// NetGINE_63471026700727
// MI455X (gfx1250) — compile-verified, matrix operand order checked by reading, not run on hardware
//
#include <hip/hip_runtime.h>
#include <hip/hip_bf16.h>
#include <stddef.h>

// ---- problem constants (match reference) ----
#define HID     256
#define NFEAT   28
#define NGRAPH  4096

typedef __attribute__((ext_vector_type(16))) __bf16 v16bf;
typedef __attribute__((ext_vector_type(8)))  __bf16 v8bf;
typedef __attribute__((ext_vector_type(8)))  float  v8f;

// ---------------- small utility kernels ----------------
__global__ void k_zero_i32(int* __restrict__ p, int n) {
  int t = blockIdx.x * blockDim.x + threadIdx.x;
  if (t < n) p[t] = 0;
}

__global__ void k_copy_i32(const int* __restrict__ a, int* __restrict__ b, int n) {
  int t = blockIdx.x * blockDim.x + threadIdx.x;
  if (t < n) b[t] = a[t];
}

__global__ void k_cvt_bf16(const float* __restrict__ in, __bf16* __restrict__ out, int n) {
  int t = blockIdx.x * blockDim.x + threadIdx.x;
  if (t < n) out[t] = (__bf16)in[t];
}

// Pack [W_rel | W_root] -> bf16 [HID, Kp] row-major, zero-padded K lanes.
__global__ void k_pack_w(const float* __restrict__ Wrel, const float* __restrict__ Wroot,
                         __bf16* __restrict__ Wp, int d, int Kp) {
  int t = blockIdx.x * blockDim.x + threadIdx.x;
  if (t >= HID * Kp) return;
  int n = t / Kp, k = t % Kp, h = Kp >> 1;
  float v = 0.0f;
  if (k < d)                    v = Wrel[n * d + k];
  else if (k >= h && k < h + d) v = Wroot[n * d + (k - h)];
  Wp[t] = (__bf16)v;
}

// ---------------- CSR build (once per launch, static graph) ----------------
__global__ void k_hist_dst(const int* __restrict__ ei, int* __restrict__ deg, int ne) {
  int t = blockIdx.x * blockDim.x + threadIdx.x;
  if (t < ne) atomicAdd(&deg[ei[ne + t]], 1);     // native global_atomic_add_u32
}

__global__ void k_hist_batch(const int* __restrict__ batch, int* __restrict__ gcnt, int nn) {
  int t = blockIdx.x * blockDim.x + threadIdx.x;
  if (t < nn) atomicAdd(&gcnt[batch[t]], 1);
}

// Single-workgroup chunked exclusive scan (Hillis-Steele in LDS, 1024/chunk).
// Writes outp[0..n-1] = exclusive prefix, outp[n] = total.
__global__ __launch_bounds__(1024) void k_scan_excl(const int* __restrict__ in,
                                                    int* __restrict__ outp, int n) {
  __shared__ int s[1024];
  __shared__ int carry;
  const int t = threadIdx.x;
  if (t == 0) carry = 0;
  __syncthreads();
  for (int base = 0; base < n; base += 1024) {
    int i = base + t;
    int v = (i < n) ? in[i] : 0;
    s[t] = v;
    __syncthreads();
    for (int off = 1; off < 1024; off <<= 1) {
      int x = (t >= off) ? s[t - off] : 0;
      __syncthreads();
      s[t] += x;
      __syncthreads();
    }
    if (i < n) outp[i] = carry + s[t] - v;
    __syncthreads();
    if (t == 0) carry += s[1023];
    __syncthreads();
  }
  if (t == 0) outp[n] = carry;
}

__global__ void k_fill_csr(const int* __restrict__ ei, int* __restrict__ cursor,
                           int* __restrict__ esrc, int ne) {
  int t = blockIdx.x * blockDim.x + threadIdx.x;
  if (t >= ne) return;
  int dd = ei[ne + t];
  int p = atomicAdd(&cursor[dd], 1);
  esrc[p] = ei[t];
}

// ---------------- fused neighbor-gather + bf16 A-pack ----------------
// One wave per node (zero atomics): agg = sum_{e in in(i)} x[src_e];
// writes Ap[i] = [bf16(agg) | pad | bf16(x_i) | pad]  (Kp columns).
__global__ __launch_bounds__(256) void k_gather_pack(
    const float* __restrict__ xin, const int* __restrict__ rowptr,
    const int* __restrict__ esrc, __bf16* __restrict__ Ap,
    int d, int Kp, int nn) {
  const int wid  = blockIdx.x * (blockDim.x >> 5) + (threadIdx.x >> 5);
  const int lane = threadIdx.x & 31;
  if (wid >= nn) return;
  const int beg = rowptr[wid], end = rowptr[wid + 1];
  const int h = Kp >> 1;
  const int per = (d + 31) >> 5;            // 1 (d=28) or 8 (d=256)
  float acc[8];
#pragma unroll 8
  for (int j = 0; j < 8; ++j) acc[j] = 0.0f;
  for (int e = beg; e < end; ++e) {
    const float* src = xin + (size_t)esrc[e] * d;
#pragma unroll 8
    for (int j = 0; j < per; ++j) {
      int c = lane + (j << 5);              // coalesced: wave covers 32 consecutive
      if (c < d) acc[j] += src[c];
    }
  }
  __bf16* arow = Ap + (size_t)wid * Kp;
  const float* self = xin + (size_t)wid * d;
#pragma unroll 8
  for (int j = 0; j < per; ++j) {
    int c = lane + (j << 5);
    if (c < d) {
      arow[c]     = (__bf16)acc[j];
      arow[h + c] = (__bf16)self[c];
    }
  }
  for (int c = d + lane; c < h; c += 32) {  // zero padding (layer 1 only)
    arow[c]     = (__bf16)0.0f;
    arow[h + c] = (__bf16)0.0f;
  }
}

// ---------------- WMMA GEMM with LDS-staged, double-buffered B ----------------
// out[M,N] = act(A[M,K](bf16) @ W[N,K](bf16)^T + bias[N]).
// Block: 256 thr = 8 waves, tile M=128 x N=64. B tile (64 rows x 32 K) staged in
// LDS once per K-chunk (8x fewer global B requests); row stride padded to 40
// elements (80B = 20 banks) so fragment ds_load_b128s are conflict-free.
#define BSTRIDE 40
__global__ __launch_bounds__(256) void k_gemm_bf16_wmma(
    const __bf16* __restrict__ A, const __bf16* __restrict__ W,
    const float* __restrict__ bias, float* __restrict__ out,
    int M, int N, int K, int do_relu) {
  __shared__ __bf16 lB[2][64 * BSTRIDE];
  const int tid  = threadIdx.x;
  const int wid  = tid >> 5;
  const int lane = tid & 31;
  const int m0   = blockIdx.x * 128 + wid * 16;
  const int n0   = blockIdx.y * 64;
  const int koff = (lane >> 4) << 3;        // 0 or 8 (CDNA5 16-bit frag layout)
  const int colb = lane & 15;
  const bool mvalid = (m0 < M);
  int rowA = m0 + colb;
  if (rowA >= M) rowA = M - 1;              // clamp; stores guarded below
  const __bf16* ap = A + (size_t)rowA * K + koff;

  // cooperative B loader: thread -> (row 0..63, 16B segment 0..3)
  const int brow = tid >> 2, bseg = (tid & 3) << 3;
  const __bf16* wsrc = W + (size_t)(n0 + brow) * K + bseg;

  v8bf breg = *(const v8bf*)(wsrc);         // prologue: chunk 0
  v8f acc[4] = {};
  const int nc = K >> 5;
  for (int c = 0; c < nc; ++c) {
    *(v8bf*)(&lB[c & 1][brow * BSTRIDE + bseg]) = breg;
    if (c + 1 < nc) breg = *(const v8bf*)(wsrc + ((c + 1) << 5));
    __syncthreads();
    const int k = c << 5;
    union { v16bf f; v8bf h[2]; } a;
    a.h[0] = *(const v8bf*)(ap + k);
    a.h[1] = *(const v8bf*)(ap + k + 16);
    __builtin_prefetch((const void*)(ap + k + 32), 0, 1);
#pragma unroll
    for (int t4 = 0; t4 < 4; ++t4) {
      union { v16bf f; v8bf h[2]; } b;
      const __bf16* lp = &lB[c & 1][(t4 * 16 + colb) * BSTRIDE + koff];
      b.h[0] = *(const v8bf*)(lp);
      b.h[1] = *(const v8bf*)(lp + 16);
      acc[t4] = __builtin_amdgcn_wmma_f32_16x16x32_bf16(
          false, a.f, false, b.f, (short)0, acc[t4], false, false);
    }
  }
  if (!mvalid) return;                      // no barriers past this point

  const int row0 = m0 + ((lane >> 4) << 3); // C layout: vgpr r -> M=r (+8 hi half)
#pragma unroll
  for (int t4 = 0; t4 < 4; ++t4) {
    int col = n0 + t4 * 16 + colb;
    float bv = bias ? bias[col] : 0.0f;
#pragma unroll
    for (int r = 0; r < 8; ++r) {
      int row = row0 + r;
      if (row < M) {
        float v = acc[t4][r] + bv;
        if (do_relu && v < 0.0f) v = 0.0f;
        out[(size_t)row * N + col] = v;
      }
    }
  }
}

// ---------------- segmented mean-pool (batch is sorted; zero atomics) ----------
__global__ __launch_bounds__(256) void k_pool_seg(
    const float* __restrict__ x1, const float* __restrict__ x2,
    const float* __restrict__ x3, const float* __restrict__ x4,
    const int* __restrict__ gstart, __bf16* __restrict__ gmb) {
  const int g = blockIdx.x;
  const int t = threadIdx.x;
  const int beg = gstart[g], end = gstart[g + 1];
  const int l = t >> 6, cloc = (t & 63) << 2;   // thread owns 4 cols of concat(1024)
  const float* xs[4] = {x1, x2, x3, x4};
  const float* xp = xs[l] + cloc;
  float4 acc = make_float4(0.f, 0.f, 0.f, 0.f);
  for (int i = beg; i < end; ++i) {
    float4 v = *(const float4*)(xp + (size_t)i * HID);
    acc.x += v.x; acc.y += v.y; acc.z += v.z; acc.w += v.w;
  }
  float inv = 1.0f / fmaxf((float)(end - beg), 1.0f);
  __bf16* o = gmb + (size_t)g * (4 * HID) + t * 4;
  o[0] = (__bf16)(acc.x * inv);
  o[1] = (__bf16)(acc.y * inv);
  o[2] = (__bf16)(acc.z * inv);
  o[3] = (__bf16)(acc.w * inv);
}

__global__ void k_fc4(const float* __restrict__ g3, const float* __restrict__ w,
                      const float* __restrict__ b, float* __restrict__ out) {
  int g = blockIdx.x * blockDim.x + threadIdx.x;
  if (g >= NGRAPH) return;
  float acc = b[0];
  const float* row = g3 + (size_t)g * HID;
#pragma unroll 8
  for (int k = 0; k < HID; ++k) acc += row[k] * w[k];
  out[g] = acc;
}

// ---------------- host-side orchestration ----------------
extern "C" void kernel_launch(void* const* d_in, const int* in_sizes, int n_in,
                              void* d_out, int out_size, void* d_ws, size_t ws_size,
                              hipStream_t stream) {
  const float* x     = (const float*)d_in[0];
  const int*   ei    = (const int*)d_in[1];     // harness integer -> int32
  const int*   batch = (const int*)d_in[2];
  const float* Wrel[4]  = {(const float*)d_in[3], (const float*)d_in[6],
                           (const float*)d_in[9], (const float*)d_in[12]};
  const float* brel[4]  = {(const float*)d_in[4], (const float*)d_in[7],
                           (const float*)d_in[10], (const float*)d_in[13]};
  const float* Wroot[4] = {(const float*)d_in[5], (const float*)d_in[8],
                           (const float*)d_in[11], (const float*)d_in[14]};
  const float* fc1_w = (const float*)d_in[15];
  const float* fc1_b = (const float*)d_in[16];
  const float* fc2_w = (const float*)d_in[17];
  const float* fc2_b = (const float*)d_in[18];
  const float* fc3_w = (const float*)d_in[19];
  const float* fc3_b = (const float*)d_in[20];
  const float* fc4_w = (const float*)d_in[21];
  const float* fc4_b = (const float*)d_in[22];
  float* out = (float*)d_out;

  const int NN = in_sizes[0] / NFEAT;   // 100000
  const int NE = in_sizes[1] / 2;       // 320000

  // bump allocator on d_ws
  char* ws = (char*)d_ws;
  size_t off = 0;
  auto alloc = [&](size_t bytes) -> void* {
    off = (off + 255) & ~(size_t)255;
    void* p = ws + off;
    off += bytes;
    return p;
  };
  int*    deg    = (int*)alloc((size_t)NN * 4);
  int*    rowptr = (int*)alloc((size_t)(NN + 1) * 4);
  int*    cursor = (int*)alloc((size_t)NN * 4);
  int*    esrc   = (int*)alloc((size_t)NE * 4);
  int*    gcnt   = (int*)alloc((size_t)NGRAPH * 4);
  int*    gstart = (int*)alloc((size_t)(NGRAPH + 1) * 4);
  __bf16* Ap     = (__bf16*)alloc((size_t)NN * 512 * 2);
  float*  X[4];
  for (int l = 0; l < 4; ++l) X[l] = (float*)alloc((size_t)NN * HID * 4);
  __bf16* Wp[4];
  Wp[0] = (__bf16*)alloc((size_t)HID * 64 * 2);
  for (int l = 1; l < 4; ++l) Wp[l] = (__bf16*)alloc((size_t)HID * 512 * 2);
  __bf16* Wfc1 = (__bf16*)alloc((size_t)HID * 1024 * 2);
  __bf16* Wfc2 = (__bf16*)alloc((size_t)HID * HID * 2);
  __bf16* Wfc3 = (__bf16*)alloc((size_t)HID * HID * 2);
  __bf16* gmb  = (__bf16*)alloc((size_t)NGRAPH * 1024 * 2);
  float*  g1   = (float*)alloc((size_t)NGRAPH * HID * 4);
  __bf16* g1b  = (__bf16*)alloc((size_t)NGRAPH * HID * 2);
  float*  g2   = (float*)alloc((size_t)NGRAPH * HID * 4);
  __bf16* g2b  = (__bf16*)alloc((size_t)NGRAPH * HID * 2);
  float*  g3   = (float*)alloc((size_t)NGRAPH * HID * 4);

  auto blocks = [](long long n) { return (unsigned)((n + 255) / 256); };

  // ---- CSR build + graph segment boundaries (once per call) ----
  k_zero_i32<<<blocks(NN), 256, 0, stream>>>(deg, NN);
  k_hist_dst<<<blocks(NE), 256, 0, stream>>>(ei, deg, NE);
  k_scan_excl<<<1, 1024, 0, stream>>>(deg, rowptr, NN);
  k_copy_i32<<<blocks(NN), 256, 0, stream>>>(rowptr, cursor, NN);
  k_fill_csr<<<blocks(NE), 256, 0, stream>>>(ei, cursor, esrc, NE);
  k_zero_i32<<<blocks(NGRAPH), 256, 0, stream>>>(gcnt, NGRAPH);
  k_hist_batch<<<blocks(NN), 256, 0, stream>>>(batch, gcnt, NN);
  k_scan_excl<<<1, 1024, 0, stream>>>(gcnt, gstart, NGRAPH);

  // ---- weights -> bf16 packed ----
  k_pack_w<<<blocks(HID * 64),  256, 0, stream>>>(Wrel[0], Wroot[0], Wp[0], NFEAT, 64);
  for (int l = 1; l < 4; ++l)
    k_pack_w<<<blocks(HID * 512), 256, 0, stream>>>(Wrel[l], Wroot[l], Wp[l], HID, 512);
  k_cvt_bf16<<<blocks(HID * 1024), 256, 0, stream>>>(fc1_w, Wfc1, HID * 1024);
  k_cvt_bf16<<<blocks(HID * HID),  256, 0, stream>>>(fc2_w, Wfc2, HID * HID);
  k_cvt_bf16<<<blocks(HID * HID),  256, 0, stream>>>(fc3_w, Wfc3, HID * HID);

  // ---- 4 GraphConv layers: gather+pack (atomic-free) then WMMA GEMM ----
  for (int l = 0; l < 4; ++l) {
    const int d  = (l == 0) ? NFEAT : HID;
    const int Kp = (l == 0) ? 64 : 512;
    const float* xin = (l == 0) ? x : X[l - 1];
    k_gather_pack<<<blocks((long long)NN * 32), 256, 0, stream>>>(
        xin, rowptr, esrc, Ap, d, Kp, NN);
    dim3 grid((NN + 127) / 128, HID / 64);
    k_gemm_bf16_wmma<<<grid, 256, 0, stream>>>(Ap, Wp[l], brel[l], X[l], NN, HID, Kp, 1);
  }

  // ---- segmented mean pool + MLP head ----
  k_pool_seg<<<NGRAPH, 256, 0, stream>>>(X[0], X[1], X[2], X[3], gstart, gmb);
  dim3 gg((NGRAPH + 127) / 128, HID / 64);
  k_gemm_bf16_wmma<<<gg, 256, 0, stream>>>(gmb, Wfc1, fc1_b, g1, NGRAPH, HID, 1024, 1);
  k_cvt_bf16<<<blocks(NGRAPH * HID), 256, 0, stream>>>(g1, g1b, NGRAPH * HID);
  k_gemm_bf16_wmma<<<gg, 256, 0, stream>>>(g1b, Wfc2, fc2_b, g2, NGRAPH, HID, HID, 1);
  k_cvt_bf16<<<blocks(NGRAPH * HID), 256, 0, stream>>>(g2, g2b, NGRAPH * HID);
  k_gemm_bf16_wmma<<<gg, 256, 0, stream>>>(g2b, Wfc3, fc3_b, g3, NGRAPH, HID, HID, 1);
  k_fc4<<<blocks(NGRAPH), 256, 0, stream>>>(g3, fc4_w, fc4_b, out);
}
